// ModConvBlock_82471962018176
// MI455X (gfx1250) — matrix operand order repaired and checked by reading, not compiled, not run on hardware
//
#include <hip/hip_runtime.h>

typedef __attribute__((ext_vector_type(16))) __bf16 v16bf;
typedef __attribute__((ext_vector_type(8)))  __bf16 v8bf;
typedef __attribute__((ext_vector_type(8)))  float  v8f;

#define NB     16
#define CIN    512
#define COUT   512
#define KS     3
#define LATENT 512
#define HH     32
#define WW     32
#define HWPIX  1024

#define AFF_SCALE   0.044194173824159216f   /* 1/sqrt(512)      */
#define CONV_SCALE  0.014731391274719739f   /* 1/sqrt(512*9)    */
#define CONV_SCALE2 (1.0f/4608.0f)          /* 1/(512*9)        */
#define LRELU_SLOPE 0.2f
#define LRELU_SCALE 1.4142135623730951f

// ---------------------------------------------------------------------------
// K1: style[n,i] = (w[n,:] . aff_w[i,:]) * aff_scale + aff_b[i]
// ---------------------------------------------------------------------------
__global__ void style_kernel(const float* __restrict__ w,
                             const float* __restrict__ aff_w,
                             const float* __restrict__ aff_b,
                             float* __restrict__ style) {
  int gid = blockIdx.x * blockDim.x + threadIdx.x;   // 16*512
  int n = gid >> 9, i = gid & 511;
  const float* wr = w + n * LATENT;
  const float* ar = aff_w + i * LATENT;
  float s = 0.f;
  #pragma unroll 4
  for (int l = 0; l < LATENT; ++l) s += wr[l] * ar[l];
  style[gid] = s * AFF_SCALE + aff_b[i];
}

// ---------------------------------------------------------------------------
// K2: wsq[o,i] = conv_scale^2 * sum_kk conv_w[o,i,kk]^2
// ---------------------------------------------------------------------------
__global__ void wsq_kernel(const float* __restrict__ conv_w,
                           float* __restrict__ wsq) {
  int gid = blockIdx.x * blockDim.x + threadIdx.x;   // 512*512
  const float* p = conv_w + (size_t)gid * 9;
  float s = 0.f;
  #pragma unroll
  for (int t = 0; t < 9; ++t) s += p[t] * p[t];
  wsq[gid] = s * CONV_SCALE2;
}

// ---------------------------------------------------------------------------
// K3: d[n,o] = rsqrt( sum_i style[n,i]^2 * wsq[o,i] + 1e-8 )
// ---------------------------------------------------------------------------
__global__ void demod_kernel(const float* __restrict__ style,
                             const float* __restrict__ wsq,
                             float* __restrict__ dmod) {
  int gid = blockIdx.x * blockDim.x + threadIdx.x;   // 16*512
  int n = gid >> 9, o = gid & 511;
  const float* sr = style + n * CIN;
  const float* qr = wsq + (size_t)o * CIN;
  float s = 0.f;
  #pragma unroll 4
  for (int i = 0; i < CIN; ++i) { float v = sr[i]; s += v * v * qr[i]; }
  dmod[gid] = rsqrtf(s + 1e-8f);
}

// ---------------------------------------------------------------------------
// K4: wb[kk][o][ci] = bf16( conv_scale * conv_w[o][ci][kk] )
// ---------------------------------------------------------------------------
__global__ void wconv_kernel(const float* __restrict__ conv_w,
                             __bf16* __restrict__ wb) {
  int gid = blockIdx.x * blockDim.x + threadIdx.x;   // 9*512*512
  int kk  = gid / (COUT * CIN);
  int rem = gid - kk * (COUT * CIN);
  int o = rem >> 9, ci = rem & 511;
  wb[gid] = (__bf16)(conv_w[((size_t)o * CIN + ci) * 9 + kk] * CONV_SCALE);
}

// ---------------------------------------------------------------------------
// K5: xb[n][ci][h][w] = bf16( style[n,ci] * input[n][ci][h][w] )
// ---------------------------------------------------------------------------
__global__ void xmod_kernel(const float* __restrict__ input,
                            const float* __restrict__ style,
                            __bf16* __restrict__ xb) {
  size_t gid = (size_t)blockIdx.x * blockDim.x + threadIdx.x; // 16*512*1024
  xb[gid] = (__bf16)(input[gid] * style[gid >> 10]);
}

// ---------------------------------------------------------------------------
// Main: implicit-GEMM 3x3 conv via V_WMMA_F32_16X16X32_BF16.
// Grid: x = pixel tile (4: rows [8t,8t+8)), y = 64-channel tile (8), z = n.
// Block: 256 threads = 8 waves in 2(M) x 4(N); wave tile 32ch x 64px.
// Per kk: 2 A-frags + 4 B-frags feed 8 WMMAs (0.75 fragment loads / WMMA).
// ---------------------------------------------------------------------------
__global__ __launch_bounds__(256, 2)
void modconv_wmma(const __bf16* __restrict__ wb,    // [9][512][512]
                  const __bf16* __restrict__ xb,    // [16][512][32][32]
                  const float*  __restrict__ dmod,  // [16][512]
                  const float*  __restrict__ noise, // [16][32][32]
                  const float*  __restrict__ noise_w,
                  const float*  __restrict__ lrelu_b,
                  float* __restrict__ out)          // [16][512][32][32]
{
  __shared__ __attribute__((aligned(16))) __bf16 A_lds[9][64][32];  // [kk][o][ci]
  __shared__ __attribute__((aligned(16))) __bf16 X_lds[10][36][32]; // [r][c][ci]

  const int tid    = threadIdx.x;
  const int lane   = tid & 31;
  const int wave   = tid >> 5;
  const int h16    = lane >> 4;   // which half of the wave
  const int l16    = lane & 15;
  const int wave_m = wave >> 2;   // 0..1 : 32-channel sub-tile
  const int wave_n = wave & 3;    // 0..3 : 64-pixel sub-tile

  const int row0   = blockIdx.x * 8;        // first output row of tile
  const int o_tile = blockIdx.y * 64;
  const int n      = blockIdx.z;

  v8f acc[2][4] = {};

  const __bf16* xb_n = xb + (size_t)n * CIN * HWPIX;

  for (int cb = 0; cb < CIN / 32; ++cb) {
    const int ci0 = cb * 32;
    __syncthreads();

    // ---- stage weights: per kk each thread moves 8 contiguous ci (b128) ----
    {
      const int o_off  = tid >> 2;         // 0..63
      const int ci_off = (tid & 3) * 8;    // 0,8,16,24
      #pragma unroll
      for (int kk = 0; kk < 9; ++kk) {
        const __bf16* src =
            wb + ((size_t)kk * COUT + o_tile + o_off) * CIN + ci0 + ci_off;
        *(v8bf*)(&A_lds[kk][o_off][ci_off]) = *(const v8bf*)src;
        if (cb + 1 < CIN / 32)                       // prefetch next ci-block
          __builtin_prefetch(src + 32, 0, 1);
      }
    }

    // ---- stage haloed activations: 32ch x 10rows x 34cols, zero padding ----
    for (int s = tid; s < 32 * 10 * 34; s += 256) {
      int ci  = s / (10 * 34);
      int rem = s - ci * (10 * 34);
      int r   = rem / 34;
      int c   = rem - r * 34;
      int row = row0 + r - 1;
      int col = c - 1;
      __bf16 v = (__bf16)0.0f;
      if ((unsigned)row < 32u && (unsigned)col < 32u)
        v = xb_n[(size_t)(ci0 + ci) * HWPIX + row * 32 + col];
      X_lds[r][c][ci] = v;
    }
    __syncthreads();

    #pragma unroll
    for (int kk = 0; kk < 9; ++kk) {
      const int ky = kk / 3, kx = kk - 3 * (kk / 3);

      // A fragments (16x32 bf16): lane l16 = row M; K runs h*8+e, 16+h*8+e
      v16bf a[2];
      #pragma unroll
      for (int mf = 0; mf < 2; ++mf) {
        const __bf16* ap = &A_lds[kk][wave_m * 32 + mf * 16 + l16][0];
        v8bf a0 = *(const v8bf*)(ap + h16 * 8);
        v8bf a1 = *(const v8bf*)(ap + 16 + h16 * 8);
        a[mf] = __builtin_shufflevector(a0, a1, 0, 1, 2, 3, 4, 5, 6, 7,
                                        8, 9, 10, 11, 12, 13, 14, 15);
      }

      #pragma unroll
      for (int nf = 0; nf < 4; ++nf) {
        // B fragment (32x16 bf16): lane l16 = col N (pixel); K runs h*8+e, 16+h*8+e
        const int p  = wave_n * 64 + nf * 16 + l16;
        const int rr = p >> 5;
        const int cc = p & 31;
        const __bf16* bp = &X_lds[rr + ky][cc + kx][h16 * 8];
        v8bf b0 = *(const v8bf*)bp;
        v8bf b1 = *(const v8bf*)(bp + 16);
        v16bf b = __builtin_shufflevector(b0, b1, 0, 1, 2, 3, 4, 5, 6, 7,
                                          8, 9, 10, 11, 12, 13, 14, 15);
        #pragma unroll
        for (int mf = 0; mf < 2; ++mf)
          acc[mf][nf] = __builtin_amdgcn_wmma_f32_16x16x32_bf16(
              false, a[mf], false, b, (short)0, acc[mf][nf], false, false);
      }
    }
  }

  // ---- epilogue: demod, noise, bias, fused leaky-ReLU * sqrt(2) ----
  const float  nw      = noise_w[0];
  const float* noise_n = noise + n * HWPIX;
  float*       out_n   = out + (size_t)n * COUT * HWPIX;

  #pragma unroll
  for (int mf = 0; mf < 2; ++mf) {
    const int o_base = o_tile + wave_m * 32 + mf * 16 + 8 * h16; // M = rr+8*h
    #pragma unroll
    for (int nf = 0; nf < 4; ++nf) {
      const int p   = wave_n * 64 + nf * 16 + l16;
      const int row = row0 + (p >> 5);
      const int col = p & 31;
      const float nz = nw * noise_n[row * 32 + col];
      #pragma unroll
      for (int rr = 0; rr < 8; ++rr) {
        const int o = o_base + rr;
        float v = acc[mf][nf][rr] * dmod[n * COUT + o] + nz + lrelu_b[o];
        v = (v >= 0.f ? v : LRELU_SLOPE * v) * LRELU_SCALE;
        out_n[(size_t)o * HWPIX + row * 32 + col] = v;
      }
    }
  }
}

// ---------------------------------------------------------------------------
extern "C" void kernel_launch(void* const* d_in, const int* in_sizes, int n_in,
                              void* d_out, int out_size, void* d_ws, size_t ws_size,
                              hipStream_t stream) {
  (void)in_sizes; (void)n_in; (void)out_size; (void)ws_size;

  const float* input   = (const float*)d_in[0];  // (16,512,32,32)
  const float* w       = (const float*)d_in[1];  // (16,512)
  const float* noise   = (const float*)d_in[2];  // (16,1,32,32)
  const float* conv_w  = (const float*)d_in[3];  // (1,512,512,3,3)
  const float* aff_w   = (const float*)d_in[4];  // (512,512)
  const float* aff_b   = (const float*)d_in[5];  // (512,)
  const float* noise_w = (const float*)d_in[6];  // (1,)
  const float* lrelu_b = (const float*)d_in[7];  // (512,)
  float* out = (float*)d_out;

  char* ws = (char*)d_ws;
  float* style = (float*)(ws + 0);                  //  32 KB
  float* dmod  = (float*)(ws + 32768);              //  32 KB
  float* wsq   = (float*)(ws + 65536);              //   1 MB
  __bf16* wb   = (__bf16*)(ws + 1114112);           // 4.5 MB  [9][512][512]
  __bf16* xb   = (__bf16*)(ws + 5832704);           //  16 MB  [16][512][1024]

  style_kernel<<<(NB * CIN) / 256, 256, 0, stream>>>(w, aff_w, aff_b, style);
  wsq_kernel<<<(COUT * CIN) / 256, 256, 0, stream>>>(conv_w, wsq);
  demod_kernel<<<(NB * COUT) / 256, 256, 0, stream>>>(style, wsq, dmod);
  wconv_kernel<<<(9 * COUT * CIN) / 256, 256, 0, stream>>>(conv_w, wb);
  xmod_kernel<<<(NB * CIN * HWPIX) / 256, 256, 0, stream>>>(input, style, xb);

  dim3 grid(4 /*pixel tiles*/, 8 /*channel tiles*/, NB);
  modconv_wmma<<<grid, 256, 0, stream>>>(wb, xb, dmod, noise, noise_w,
                                         lrelu_b, out);
}
